// OuterMean_14190571946043
// MI455X (gfx1250) — matrix operand order repaired and checked by reading, not compiled, not run on hardware
//
#include <hip/hip_runtime.h>

// ---------------------------------------------------------------------------
// OuterProductMean for MI455X (gfx1250, wave32, WMMA bf16 16x16x32)
// B=1, M=128, I=512, D=128
// ---------------------------------------------------------------------------

#define M_DEPTH 128
#define I_DIM   512
#define D_DIM   128
#define EPS     1e-5f
#define LN_EPS  1e-5f

typedef __attribute__((ext_vector_type(16))) __bf16 v16bf;
typedef __attribute__((ext_vector_type(8)))  float  v8f;

union Frag16 {
    v16bf v;
    uint4 q[2];     // 2 x 16B = 32B = 16 bf16
};

__device__ __forceinline__ unsigned short f2bf(float f) {
    unsigned u = __builtin_bit_cast(unsigned, f);
    u += 0x7FFFu + ((u >> 16) & 1u);          // round-to-nearest-even
    return (unsigned short)(u >> 16);
}

// ---------------------------------------------------------------------------
// Convert weights fp32 -> bf16, transposed to N-major (wT[dout][din]) so the
// WMMA B-fragment (lane n holds two runs of 8 contiguous K values) loads as uint4 pairs.
// ---------------------------------------------------------------------------
__global__ __launch_bounds__(256) void k_prep_w(
    const float* __restrict__ wl, const float* __restrict__ wr,
    const float* __restrict__ wo,
    unsigned short* __restrict__ wlT, unsigned short* __restrict__ wrT,
    unsigned short* __restrict__ woT)
{
    int idx = blockIdx.x * 256 + threadIdx.x;          // din*128 + dout
    if (idx >= D_DIM * D_DIM) return;
    int din = idx >> 7, dout = idx & 127;
    int t = dout * D_DIM + din;
    wlT[t] = f2bf(wl[idx]);
    wrT[t] = f2bf(wr[idx]);
    woT[t] = f2bf(wo[idx]);
}

// ---------------------------------------------------------------------------
// cntinv[i][j] = 1 / (M * (sum_m mask[m,i]*mask[m,j] + eps))
// ---------------------------------------------------------------------------
__global__ __launch_bounds__(256) void k_cnt(
    const unsigned char* __restrict__ mask, float* __restrict__ cntinv)
{
    int idx = blockIdx.x * 256 + threadIdx.x;          // i*512 + j
    int i = idx >> 9, j = idx & 511;
    int c = 0;
    #pragma unroll 8
    for (int m = 0; m < M_DEPTH; ++m)
        c += (int)((mask[m * I_DIM + i] & mask[m * I_DIM + j]) & 1u);
    cntinv[idx] = 1.0f / (((float)c + EPS) * (float)M_DEPTH);
}

// ---------------------------------------------------------------------------
// Stage 1: LayerNorm + left/right projections + mask, scattered to
// Lws/Rws[d][i][m] (m innermost => K-dim of the stage-2 contraction).
// Block = 256 threads (8 waves), handles 64 rows of xn.
// ---------------------------------------------------------------------------
__global__ __launch_bounds__(256) void k_ln_proj(
    const float* __restrict__ x, const unsigned char* __restrict__ mask,
    const float* __restrict__ ln_g, const float* __restrict__ ln_b,
    const unsigned short* __restrict__ wlT, const float* __restrict__ bl,
    const unsigned short* __restrict__ wrT, const float* __restrict__ br,
    unsigned short* __restrict__ Lws, unsigned short* __restrict__ Rws)
{
    __shared__ unsigned short xn[64][136];   // +8 shorts pad (bank stagger)
    __shared__ float mrow[64];

    const int tid  = threadIdx.x;
    const int lane = tid & 31;
    const int wave = tid >> 5;
    const int row0 = blockIdx.x * 64;

    // ---- LayerNorm: each wave normalizes 8 rows (32 lanes x 4 floats) ----
    for (int rr = 0; rr < 8; ++rr) {
        const int rloc = wave * 8 + rr;
        const int row  = row0 + rloc;
        float4 v = *(const float4*)(x + (size_t)row * D_DIM + lane * 4);
        float s  = v.x + v.y + v.z + v.w;
        float s2 = v.x * v.x + v.y * v.y + v.z * v.z + v.w * v.w;
        #pragma unroll
        for (int off = 16; off > 0; off >>= 1) {
            s  += __shfl_xor(s,  off, 32);
            s2 += __shfl_xor(s2, off, 32);
        }
        float mu  = s * (1.0f / D_DIM);
        float var = s2 * (1.0f / D_DIM) - mu * mu;
        float rs  = rsqrtf(var + LN_EPS);
        float4 g = *(const float4*)(ln_g + lane * 4);
        float4 b = *(const float4*)(ln_b + lane * 4);
        ushort4 o;
        o.x = f2bf((v.x - mu) * rs * g.x + b.x);
        o.y = f2bf((v.y - mu) * rs * g.y + b.y);
        o.z = f2bf((v.z - mu) * rs * g.z + b.z);
        o.w = f2bf((v.w - mu) * rs * g.w + b.w);
        *(ushort4*)(&xn[rloc][lane * 4]) = o;
        if (lane == 0) mrow[rloc] = mask[row] ? 1.0f : 0.0f;
    }
    __syncthreads();

    // ---- GEMM: 64 tiles (4 row-tiles x 8 d-tiles x {L,R}); 8 per wave ----
    const int h  = lane >> 4;      // lane half
    const int il = lane & 15;
    for (int s = 0; s < 8; ++s) {
        const int tile = s * 8 + wave;
        const int isR  = tile >> 5;
        const int tt   = tile & 31;
        const int t    = tt >> 3;                  // row tile 0..3
        const int n    = tt & 7;                   // d tile 0..7
        const unsigned short* wT  = isR ? wrT : wlT;
        const float*          bia = isR ? br  : bl;
        unsigned short*       dst = isR ? Rws : Lws;

        v8f acc = {};
        #pragma unroll
        for (int kk = 0; kk < D_DIM; kk += 32) {
            Frag16 a, bfr;
            // A frag: lane row = t*16+il; K runs {kk+8h..+7, kk+16+8h..+7}
            const unsigned short* ap = &xn[t * 16 + il][kk + 8 * h];
            a.q[0] = *(const uint4*)(ap);
            a.q[1] = *(const uint4*)(ap + 16);
            // B frag: lane col = n*16+il; K runs {kk+8h..+7, kk+16+8h..+7} (N-major weights)
            const unsigned short* bp = wT + (size_t)(n * 16 + il) * D_DIM + kk + 8 * h;
            bfr.q[0] = *(const uint4*)(bp);
            bfr.q[1] = *(const uint4*)(bp + 16);
            acc = __builtin_amdgcn_wmma_f32_16x16x32_bf16(
                false, a.v, false, bfr.v, (short)0, acc, false, false);
        }

        const float bn = bia[n * 16 + il];
        #pragma unroll
        for (int r = 0; r < 8; ++r) {
            const int rloc = t * 16 + 8 * h + r;   // D-frag row layout
            const float vv = (acc[r] + bn) * mrow[rloc];
            const int grow = row0 + rloc;
            const int m = grow >> 9;               // row = m*512 + i
            const int i = grow & 511;
            const int d = n * 16 + il;
            dst[((size_t)d * I_DIM + i) * M_DEPTH + m] = f2bf(vv);
        }
    }
}

// ---------------------------------------------------------------------------
// Stage 2+3 fused: per 16x16 (i,j) tile:
//   phase A: outer[i,j,d] = (sum_m L*R) * cntinv  -> bf16 panel in LDS
//   phase B: out[i,j,:]   = panel @ woT + bo      -> fp32 to d_out
// Block = 256 threads (8 waves). Dynamic LDS = 1KB cnt + 256x136 bf16 panel.
// ---------------------------------------------------------------------------
__global__ __launch_bounds__(256) void k_outer_proj(
    const unsigned short* __restrict__ Lws, const unsigned short* __restrict__ Rws,
    const float* __restrict__ cntinv,
    const unsigned short* __restrict__ woT, const float* __restrict__ bo,
    float* __restrict__ out)
{
    extern __shared__ char smem[];
    float*          cnt_s = (float*)smem;                       // 256 f32
    unsigned short* Apan  = (unsigned short*)(smem + 1024);     // 256 x 136 bf16

    const int tid  = threadIdx.x;
    const int lane = tid & 31;
    const int wave = tid >> 5;
    const int i0 = blockIdx.y * 16;
    const int j0 = blockIdx.x * 16;
    const int h  = lane >> 4;
    const int il = lane & 15;

    cnt_s[tid] = cntinv[(size_t)(i0 + (tid >> 4)) * I_DIM + j0 + (tid & 15)];
    __syncthreads();

    // ---- phase A: wave handles d = wave*16 .. wave*16+15 ----
    for (int dd = 0; dd < 16; ++dd) {
        const int d = wave * 16 + dd;
        const unsigned short* Lp = Lws + (size_t)d * I_DIM * M_DEPTH;
        const unsigned short* Rp = Rws + (size_t)d * I_DIM * M_DEPTH;
        v8f acc = {};
        #pragma unroll
        for (int kk = 0; kk < M_DEPTH; kk += 32) {
            Frag16 a, bfr;
            const unsigned short* ap = Lp + (size_t)(i0 + il) * M_DEPTH + kk + 8 * h;
            a.q[0] = *(const uint4*)(ap);
            a.q[1] = *(const uint4*)(ap + 16);
            const unsigned short* bp = Rp + (size_t)(j0 + il) * M_DEPTH + kk + 8 * h;
            bfr.q[0] = *(const uint4*)(bp);
            bfr.q[1] = *(const uint4*)(bp + 16);
            acc = __builtin_amdgcn_wmma_f32_16x16x32_bf16(
                false, a.v, false, bfr.v, (short)0, acc, false, false);
        }
        #pragma unroll
        for (int r = 0; r < 8; ++r) {
            const int iloc = 8 * h + r;
            const float vv = acc[r] * cnt_s[iloc * 16 + il];
            Apan[(size_t)(iloc * 16 + il) * 136 + d] = f2bf(vv);
        }
    }
    __syncthreads();

    // ---- phase B: 256 (i,j) rows x 128 dout = 16 row-tiles x 8 n-tiles ----
    for (int s = 0; s < 16; ++s) {
        const int tile = s * 8 + wave;
        const int t = tile >> 3;     // row tile 0..15
        const int n = tile & 7;      // dout tile 0..7
        v8f acc = {};
        #pragma unroll
        for (int kk = 0; kk < D_DIM; kk += 32) {
            Frag16 a, bfr;
            const unsigned short* ap = Apan + (size_t)(t * 16 + il) * 136 + kk + 8 * h;
            a.q[0] = *(const uint4*)(ap);
            a.q[1] = *(const uint4*)(ap + 16);
            const unsigned short* bp = woT + (size_t)(n * 16 + il) * D_DIM + kk + 8 * h;
            bfr.q[0] = *(const uint4*)(bp);
            bfr.q[1] = *(const uint4*)(bp + 16);
            acc = __builtin_amdgcn_wmma_f32_16x16x32_bf16(
                false, a.v, false, bfr.v, (short)0, acc, false, false);
        }
        const float bn = bo[n * 16 + il];
        #pragma unroll
        for (int r = 0; r < 8; ++r) {
            const int rowl = t * 16 + 8 * h + r;   // flattened (iloc, jloc)
            const int iloc = rowl >> 4;
            const int jloc = rowl & 15;
            out[((size_t)(i0 + iloc) * I_DIM + (j0 + jloc)) * D_DIM + n * 16 + il]
                = acc[r] + bn;
        }
    }
}

// ---------------------------------------------------------------------------
extern "C" void kernel_launch(void* const* d_in, const int* in_sizes, int n_in,
                              void* d_out, int out_size, void* d_ws, size_t ws_size,
                              hipStream_t stream)
{
    const float*         x    = (const float*)d_in[0];
    const unsigned char* mask = (const unsigned char*)d_in[1];   // jax bool = 1B
    const float*         ln_g = (const float*)d_in[2];
    const float*         ln_b = (const float*)d_in[3];
    const float*         wl   = (const float*)d_in[4];
    const float*         bl   = (const float*)d_in[5];
    const float*         wr   = (const float*)d_in[6];
    const float*         br   = (const float*)d_in[7];
    const float*         wo   = (const float*)d_in[8];
    const float*         bo   = (const float*)d_in[9];
    float*               outp = (float*)d_out;

    // Workspace layout (bytes)
    char* ws = (char*)d_ws;
    unsigned short* wlT    = (unsigned short*)(ws + 0);             // 32 KB
    unsigned short* wrT    = (unsigned short*)(ws + 32768);         // 32 KB
    unsigned short* woT    = (unsigned short*)(ws + 65536);         // 32 KB
    float*          cntinv = (float*)(ws + 98304);                  // 1 MB
    unsigned short* Lws    = (unsigned short*)(ws + 1146880);       // 16 MB
    unsigned short* Rws    = (unsigned short*)(ws + 17924096);      // 16 MB

    k_prep_w<<<(D_DIM * D_DIM + 255) / 256, 256, 0, stream>>>(wl, wr, wo, wlT, wrT, woT);
    k_cnt<<<(I_DIM * I_DIM) / 256, 256, 0, stream>>>(mask, cntinv);
    k_ln_proj<<<(M_DEPTH * I_DIM) / 64, 256, 0, stream>>>(
        x, mask, ln_g, ln_b, wlT, bl, wrT, br, Lws, Rws);

    dim3 grid2(I_DIM / 16, I_DIM / 16);
    size_t smem2 = 1024 + (size_t)256 * 136 * sizeof(unsigned short);
    k_outer_proj<<<grid2, 256, smem2, stream>>>(Lws, Rws, cntinv, woT, bo, outp);
}
